// PU_plus_NTXent_30185030156996
// MI455X (gfx1250) — matrix operand order repaired and checked by reading, not compiled, not run on hardware
//
#include <hip/hip_runtime.h>
#include <hip/hip_bf16.h>
#include <math.h>

// ---------------------------------------------------------------------------
// PU + NT-Xent fused loss for MI455X (gfx1250, wave32, WMMA).
//
// Heavy work = two GEMMs (4096x512x2048 MLP, 4096x4096x256 similarity),
// both on v_wmma_f32_16x16x32_f16 with 1 A-frag x 4 C-tile register
// blocking (4x A-reuse). PU loss reduced analytically to scalar sums over
// pred[4096]. NT-Xent logsumexp uses fixed max 2.0 (= 1/TEMP bound).
// ---------------------------------------------------------------------------

typedef __attribute__((ext_vector_type(16))) _Float16 v16h;
typedef __attribute__((ext_vector_type(8)))  _Float16 v8h;
typedef __attribute__((ext_vector_type(8)))  float    v8f;

#define BATCH   2048
#define NROWS   4096      // 2*BATCH
#define LATENT  2048
#define HIDDEN  512
#define ZDIM    256
#define INV_TEMP 2.0f     // 1/0.5
#define PRIOR       0.3f
#define PRIOR_PRIME 0.5f

// ---- workspace layout (bytes) ----------------------------------------------
#define WS_X    0                               // f16 [4096][2048]  16 MiB
#define WS_W1   (WS_X  + NROWS*LATENT*2)        // f16 [512][2048]    2 MiB
#define WS_ZN   (WS_W1 + HIDDEN*LATENT*2)       // f16 [4096][256]    2 MiB
#define WS_PRED (WS_ZN + NROWS*ZDIM*2)          // f32 [4096]
#define WS_ACC  (WS_PRED + NROWS*4)             // f32 [0]=ntxent sum, [1]=onnpu

// ---------------------------------------------------------------------------
// Fragment loaders for row-major f16 storage (contiguous k order, same for A and B).
// A (16x32, MxK): lane L: row = lr (=L%16); h (=L/16) picks K-halves:
//   elems 0..15 = K[kBase + h*16 .. +15]  (matches B's contiguous order)
// B (32x16, KxN) from row-major S[N][K]: lane L: col n = nBase+lr,
//   elems 0..15 = S[n][kBase + h*16 .. +15]  (one contiguous 32B load)
// ---------------------------------------------------------------------------
__device__ __forceinline__ v16h load_frag_a(const _Float16* rowPtr, int kBase, int h) {
  v8h a0 = *(const v8h*)(rowPtr + kBase + h * 16);
  v8h a1 = *(const v8h*)(rowPtr + kBase + h * 16 + 8);
  v16h r;
#pragma unroll
  for (int i = 0; i < 8; ++i) { r[i] = a0[i]; r[i + 8] = a1[i]; }
  return r;
}

__device__ __forceinline__ v16h load_frag_b(const _Float16* S, int n, int ld,
                                            int kBase, int h) {
  return *(const v16h*)(S + (size_t)n * ld + kBase + h * 16);
}

__device__ __forceinline__ v8f wmma_f16(v16h a, v16h b, v8f c) {
  return __builtin_amdgcn_wmma_f32_16x16x32_f16(false, a, false, b,
                                                (short)0, c, false, false);
}

// ---------------------------------------------------------------------------
// 0) zero the accumulators (graph-replay safe re-init each launch)
// ---------------------------------------------------------------------------
__global__ void k_init(float* acc) {
  if (threadIdx.x < 2) acc[threadIdx.x] = 0.0f;
}

// ---------------------------------------------------------------------------
// 1) convert concat(h_i,h_j) [4096][2048] f32 -> f16
// ---------------------------------------------------------------------------
__global__ void k_convert_x(const float* __restrict__ hi,
                            const float* __restrict__ hj,
                            _Float16* __restrict__ X) {
  int t   = blockIdx.x * blockDim.x + threadIdx.x;   // 4096*2048/8 threads
  int row = t >> 8;                                  // 256 threads per row
  int col = (t & 255) * 8;
  const float* src = (row < BATCH) ? hi + (size_t)row * LATENT
                                   : hj + (size_t)(row - BATCH) * LATENT;
  float4 a = *(const float4*)(src + col);
  float4 b = *(const float4*)(src + col + 4);
  v8h o;
  o[0] = (_Float16)a.x; o[1] = (_Float16)a.y; o[2] = (_Float16)a.z; o[3] = (_Float16)a.w;
  o[4] = (_Float16)b.x; o[5] = (_Float16)b.y; o[6] = (_Float16)b.z; o[7] = (_Float16)b.w;
  *(v8h*)(X + (size_t)row * LATENT + col) = o;
}

// ---------------------------------------------------------------------------
// 2) convert W1 [512][2048] f32 -> f16 (row-major kept: B[k][n]=W1[n][k])
// ---------------------------------------------------------------------------
__global__ void k_convert_w1(const float* __restrict__ W1, _Float16* __restrict__ W) {
  int t = blockIdx.x * blockDim.x + threadIdx.x;     // 512*2048/8 threads
  int base = t * 8;
  float4 a = *(const float4*)(W1 + base);
  float4 b = *(const float4*)(W1 + base + 4);
  v8h o;
  o[0] = (_Float16)a.x; o[1] = (_Float16)a.y; o[2] = (_Float16)a.z; o[3] = (_Float16)a.w;
  o[4] = (_Float16)b.x; o[5] = (_Float16)b.y; o[6] = (_Float16)b.z; o[7] = (_Float16)b.w;
  *(v8h*)(W + base) = o;
}

// ---------------------------------------------------------------------------
// 3) row-normalize concat(z_i,z_j) [4096][256] -> f16 Zn.  One wave per row.
// ---------------------------------------------------------------------------
__global__ void k_norm_z(const float* __restrict__ zi, const float* __restrict__ zj,
                         _Float16* __restrict__ Zn) {
  int wave = (blockIdx.x * blockDim.x + threadIdx.x) >> 5;   // row
  int lane = threadIdx.x & 31;
  const float* src = (wave < BATCH) ? zi + (size_t)wave * ZDIM
                                    : zj + (size_t)(wave - BATCH) * ZDIM;
  float4 a = *(const float4*)(src + lane * 8);
  float4 b = *(const float4*)(src + lane * 8 + 4);
  float ss = a.x*a.x + a.y*a.y + a.z*a.z + a.w*a.w
           + b.x*b.x + b.y*b.y + b.z*b.z + b.w*b.w;
#pragma unroll
  for (int off = 1; off < 32; off <<= 1) ss += __shfl_xor(ss, off, 32);
  float inv = 1.0f / fmaxf(sqrtf(ss), 1e-8f);
  v8h o;
  o[0] = (_Float16)(a.x*inv); o[1] = (_Float16)(a.y*inv);
  o[2] = (_Float16)(a.z*inv); o[3] = (_Float16)(a.w*inv);
  o[4] = (_Float16)(b.x*inv); o[5] = (_Float16)(b.y*inv);
  o[6] = (_Float16)(b.z*inv); o[7] = (_Float16)(b.w*inv);
  *(v8h*)(Zn + (size_t)wave * ZDIM + lane * 8) = o;
}

// ---------------------------------------------------------------------------
// 4) MLP: pred[r] = W2 . LeakyReLU(X[r]@W1^T + b1) + b2.  One wave / 16 rows.
//    Register blocking: 1 A fragment feeds 4 C tiles (4 hidden tiles).
// ---------------------------------------------------------------------------
__global__ void __launch_bounds__(128)
k_mlp_pred(const _Float16* __restrict__ X, const _Float16* __restrict__ W1,
           const float* __restrict__ b1, const float* __restrict__ W2,
           const float* __restrict__ b2, float* __restrict__ pred) {
  int wave = (blockIdx.x * blockDim.x + threadIdx.x) >> 5;   // 0..255
  int lane = threadIdx.x & 31;
  int lr = lane & 15, h = lane >> 4;
  int rowBase = wave * 16;
  const _Float16* arow = X + (size_t)(rowBase + lr) * LATENT;

  float acc[8] = {0.f,0.f,0.f,0.f,0.f,0.f,0.f,0.f};
  for (int ntg = 0; ntg < HIDDEN / 64; ++ntg) {     // 8 groups of 4 tiles
    int n0 = ntg * 64 + lr;                         // column of tile j: n0+16j
    v8f c[4] = {{}, {}, {}, {}};
#pragma unroll 2
    for (int kt = 0; kt < LATENT / 32; ++kt) {      // 64 K steps
      v16h a = load_frag_a(arow, kt * 32, h);       // 1 A load ...
#pragma unroll
      for (int j = 0; j < 4; ++j) {                 // ... reused by 4 wmma
        v16h b = load_frag_b(W1, n0 + j * 16, LATENT, kt * 32, h);
        c[j] = wmma_f16(a, b, c[j]);
      }
    }
#pragma unroll
    for (int j = 0; j < 4; ++j) {
      int n = n0 + j * 16;
      float bias = b1[n], w2n = W2[n];
#pragma unroll
      for (int p = 0; p < 8; ++p) {
        float v = c[j][p] + bias;
        v = (v >= 0.0f) ? v : 0.01f * v;            // LeakyReLU(0.01)
        acc[p] += v * w2n;
      }
    }
  }
  // reduce over the 16 lanes of each C half (rows p / 8+p)
#pragma unroll
  for (int p = 0; p < 8; ++p) {
    float v = acc[p];
#pragma unroll
    for (int off = 1; off < 16; off <<= 1) v += __shfl_xor(v, off, 32);
    acc[p] = v;
  }
  if (lr == 0) {
#pragma unroll
    for (int p = 0; p < 8; ++p)
      pred[rowBase + p + 8 * h] = acc[p] + b2[0];
  }
}

// ---------------------------------------------------------------------------
// 5) NT-Xent: one wave / 16 rows. sim[r,c] = (zn_r . zn_c)/TEMP via WMMA,
//    1 A-frag x 4 column tiles, fixed-max logsumexp (bound 2.0), diagonal
//    masked, positive captured from its C tile.  Sum (lse - pos) atomically.
// ---------------------------------------------------------------------------
__global__ void __launch_bounds__(128)
k_ntxent(const _Float16* __restrict__ Zn, float* __restrict__ acc) {
  int wave = (blockIdx.x * blockDim.x + threadIdx.x) >> 5;   // 0..255
  int lane = threadIdx.x & 31;
  int lr = lane & 15, h = lane >> 4;
  int rowBase = wave * 16;
  int diagTile = rowBase >> 4;
  int partTile = (rowBase < BATCH) ? diagTile + (BATCH >> 4)
                                   : diagTile - (BATCH >> 4);
  const _Float16* arow = Zn + (size_t)(rowBase + lr) * ZDIM;

  float s[8]   = {0.f,0.f,0.f,0.f,0.f,0.f,0.f,0.f};
  float pos[8] = {0.f,0.f,0.f,0.f,0.f,0.f,0.f,0.f};

  for (int ctg = 0; ctg < NROWS / 64; ++ctg) {      // 64 groups of 4 tiles
    v8f c[4] = {{}, {}, {}, {}};
#pragma unroll
    for (int kt = 0; kt < ZDIM / 32; ++kt) {        // 8 K steps
      v16h a = load_frag_a(arow, kt * 32, h);       // 1 A load ...
#pragma unroll
      for (int j = 0; j < 4; ++j) {                 // ... reused by 4 wmma
        v16h b = load_frag_b(Zn, ctg * 64 + j * 16 + lr, ZDIM, kt * 32, h);
        c[j] = wmma_f16(a, b, c[j]);
      }
    }
#pragma unroll
    for (int j = 0; j < 4; ++j) {
      int ct = ctg * 4 + j;
      bool isDiag = (ct == diagTile);
      bool isPart = (ct == partTile);
#pragma unroll
      for (int p = 0; p < 8; ++p) {
        float v = INV_TEMP * c[j][p];               // sim value
        bool mine = (lr == p + 8 * h);              // col offset == row%16
        pos[p] += (isPart && mine) ? v : 0.0f;      // capture positive
        float e = __expf(v - 2.0f);                 // fixed max = 2.0
        s[p] += (isDiag && mine) ? 0.0f : e;        // mask c == r
      }
    }
  }
  float wsum = 0.0f;
#pragma unroll
  for (int p = 0; p < 8; ++p) {
    float sv = s[p], pv = pos[p];
#pragma unroll
    for (int off = 1; off < 16; off <<= 1) {
      sv += __shfl_xor(sv, off, 32);
      pv += __shfl_xor(pv, off, 32);
    }
    wsum += (2.0f + __logf(sv)) - pv;               // lse - positive
  }
  if (lr == 0) atomicAdd(&acc[0], wsum);            // lanes 0 & 16: 8 rows each
}

// ---------------------------------------------------------------------------
// 6) nnPU loss: broadcasting collapses to scalar sums of sigmoid(+-pred).
// ---------------------------------------------------------------------------
__global__ void __launch_bounds__(1024)
k_pu(const float* __restrict__ pred, const int* __restrict__ target,
     float* __restrict__ acc) {
  __shared__ float shS[32], shN[32], shP[32];
  float sp = 0.f, sn = 0.f, np = 0.f;
  for (int i = threadIdx.x; i < NROWS; i += blockDim.x) {
    float x  = pred[i];
    float sg = 1.0f / (1.0f + __expf(-x));          // sigmoid(x)
    sp += sg;                                       // sum sigmoid(pred)
    sn += 1.0f - sg;                                // sum sigmoid(-pred)
    np += (float)target[i & (BATCH - 1)];           // concat(target,target)
  }
#pragma unroll
  for (int off = 1; off < 32; off <<= 1) {
    sp += __shfl_xor(sp, off, 32);
    sn += __shfl_xor(sn, off, 32);
    np += __shfl_xor(np, off, 32);
  }
  int wid = threadIdx.x >> 5;
  if ((threadIdx.x & 31) == 0) { shS[wid] = sp; shN[wid] = sn; shP[wid] = np; }
  __syncthreads();
  if (threadIdx.x == 0) {
    float S2 = 0.f, S1 = 0.f, Np = 0.f;
    int nw = blockDim.x >> 5;
    for (int w = 0; w < nw; ++w) { S2 += shS[w]; S1 += shN[w]; Np += shP[w]; }
    float Nu = (float)NROWS - Np;
    float n_pos = fmaxf(1.0f, Np), n_unl = fmaxf(1.0f, Nu);
    float positive_risk = PRIOR_PRIME / n_pos * (Np * S1);
    float negative_risk =
        (1.0f - PRIOR_PRIME) / (n_unl * (1.0f - PRIOR)) * (Nu * S2)
      - (1.0f - PRIOR_PRIME) * PRIOR / (n_pos * (1.0f - PRIOR)) * (Np * S2);
    float n = (float)NROWS;
    acc[1] = (negative_risk < 0.0f) ? (-negative_risk / n)
                                    : ((positive_risk + negative_risk) / n);
  }
}

// ---------------------------------------------------------------------------
// 7) combine
// ---------------------------------------------------------------------------
__global__ void k_finalize(const float* __restrict__ acc, float* __restrict__ out) {
  out[0] = acc[1] + acc[0] / (float)NROWS;
}

// ---------------------------------------------------------------------------
extern "C" void kernel_launch(void* const* d_in, const int* in_sizes, int n_in,
                              void* d_out, int out_size, void* d_ws, size_t ws_size,
                              hipStream_t stream) {
  const float* h_i = (const float*)d_in[0];
  const float* h_j = (const float*)d_in[1];
  const float* z_i = (const float*)d_in[2];
  const float* z_j = (const float*)d_in[3];
  const float* W1  = (const float*)d_in[4];
  const float* b1  = (const float*)d_in[5];
  const float* W2  = (const float*)d_in[6];
  const float* b2  = (const float*)d_in[7];
  const int*   tgt = (const int*)d_in[8];

  char* ws = (char*)d_ws;
  _Float16* Xf   = (_Float16*)(ws + WS_X);
  _Float16* W1f  = (_Float16*)(ws + WS_W1);
  _Float16* Znf  = (_Float16*)(ws + WS_ZN);
  float*    pred = (float*)(ws + WS_PRED);
  float*    acc  = (float*)(ws + WS_ACC);
  float*    out  = (float*)d_out;

  k_init<<<1, 64, 0, stream>>>(acc);
  k_convert_x <<<NROWS * LATENT / 8 / 256, 256, 0, stream>>>(h_i, h_j, Xf);
  k_convert_w1<<<HIDDEN * LATENT / 8 / 256, 256, 0, stream>>>(W1, W1f);
  k_norm_z    <<<NROWS / 8, 256, 0, stream>>>(z_i, z_j, Znf);
  k_mlp_pred  <<<NROWS / 16 / 4, 128, 0, stream>>>(Xf, W1f, b1, W2, b2, pred);
  k_ntxent    <<<NROWS / 16 / 4, 128, 0, stream>>>(Znf, acc);
  k_pu        <<<1, 1024, 0, stream>>>(pred, tgt, acc);
  k_finalize  <<<1, 1, 0, stream>>>(acc, out);
}
